// multihead_self_attention_14766097563710
// MI455X (gfx1250) — matrix operand order repaired and checked by reading, not compiled, not run on hardware
//
#include <hip/hip_runtime.h>

// ---------------------------------------------------------------------------
// MI455X (gfx1250) multi-head self-attention with RoPE, causal, softmax.
// bf16 WMMA (v_wmma_f32_16x16x32_bf16), fp32 accumulation.
// Round 3: ping-pong (unroll-by-2) software pipeline in the GEMM k-loop to
// eliminate the v_dual_mov register-rotation overhead seen in round 2.
// ---------------------------------------------------------------------------

typedef __attribute__((ext_vector_type(16))) __bf16 bf16x16;
typedef __attribute__((ext_vector_type(8)))  __bf16 bf16x8;
typedef __attribute__((ext_vector_type(8)))  float  f32x8;

constexpr int BATCH  = 2;
constexpr int SEQ    = 2048;
constexpr int DMODEL = 768;
constexpr int NHEAD  = 12;
constexpr int DK     = 64;
constexpr int MTOT   = BATCH * SEQ;   // 4096 total token rows

__device__ __forceinline__ f32x8 wmma_bf16(bf16x16 a, bf16x16 b, f32x8 c) {
    return __builtin_amdgcn_wmma_f32_16x16x32_bf16(
        /*neg_a=*/false, a, /*neg_b=*/false, b,
        /*c_mod=*/(short)0, c, /*reuse_a=*/false, /*reuse_b=*/false);
}

// A fragment, 16x32 (MxK). Lane row = lane&15; lane-half selects K chunks:
// half 0: K[0..15], half 1: K[16..31]. Two 16B loads.
__device__ __forceinline__ bf16x16 load_a_frag(const __bf16* rowptr, int hf) {
    bf16x8 lo = *(const bf16x8*)(rowptr + 16 * hf);
    bf16x8 hi = *(const bf16x8*)(rowptr + 16 * hf + 8);
    bf16x16 r;
#pragma unroll
    for (int i = 0; i < 8; ++i) { r[i] = lo[i]; r[i + 8] = hi[i]; }
    return r;
}

// B fragment, 32x16 (KxN). Lane col = lane&15; lane-half selects K 0..15 vs
// 16..31; each lane reads 16 consecutive K values (two 16B loads).
__device__ __forceinline__ bf16x16 load_b_frag(const __bf16* p) {
    bf16x8 lo = *(const bf16x8*)p;
    bf16x8 hi = *(const bf16x8*)(p + 8);
    bf16x16 r;
#pragma unroll
    for (int i = 0; i < 8; ++i) { r[i] = lo[i]; r[i + 8] = hi[i]; }
    return r;
}

// Fragment set for one 32-wide k-slice of a 32(M)x64(N) wave tile.
struct KSlice {
    bf16x16 a0, a1;
    bf16x16 b[4];
};

__device__ __forceinline__ void load_slice(KSlice& s, const __bf16* arow0,
                                           const __bf16* arow1, const __bf16* wbase,
                                           int k, int hf) {
    s.a0 = load_a_frag(arow0 + k, hf);
    s.a1 = load_a_frag(arow1 + k, hf);
#pragma unroll
    for (int nt = 0; nt < 4; ++nt)
        s.b[nt] = load_b_frag(wbase + k + (size_t)(nt * 16) * DMODEL);
}

__device__ __forceinline__ void compute_slice(f32x8 acc[2][4], const KSlice& s) {
#pragma unroll
    for (int nt = 0; nt < 4; ++nt) {
        acc[0][nt] = wmma_bf16(s.a0, s.b[nt], acc[0][nt]);
        acc[1][nt] = wmma_bf16(s.a1, s.b[nt], acc[1][nt]);
    }
}

// ---------------------------------------------------------------------------
// 0) fp32 -> bf16 conversion (vectorized float4 loads)
// ---------------------------------------------------------------------------
__global__ __launch_bounds__(256) void cvt_f32_to_bf16(const float* __restrict__ src,
                                                       __bf16* __restrict__ dst, int n) {
    int i = (blockIdx.x * 256 + threadIdx.x) * 4;
    if (i >= n) return;
    float4 v = *(const float4*)(src + i);
    dst[i + 0] = (__bf16)v.x;
    dst[i + 1] = (__bf16)v.y;
    dst[i + 2] = (__bf16)v.z;
    dst[i + 3] = (__bf16)v.w;
}

// ---------------------------------------------------------------------------
// 1) QKV projection: Out[b,h,s,dk] = X[t,:] . W[o,:]  (NT GEMM, o = h*64+dk)
//    Block = 8 waves; wave tile 32(M) x 64(N); block tile 256 x 64.
//    Ping-pong pipelined k-loop: no register rotation copies.
// ---------------------------------------------------------------------------
__global__ __launch_bounds__(256) void gemm_qkv_heads(const __bf16* __restrict__ X,
                                                      const __bf16* __restrict__ W,
                                                      __bf16* __restrict__ Out) {
    int wave = threadIdx.x >> 5, lane = threadIdx.x & 31;
    int ln = lane & 15, hf = lane >> 4;
    int m0 = blockIdx.x * 256 + wave * 32;
    int h  = blockIdx.y;

    const __bf16* arow0 = X + (size_t)(m0 + ln) * DMODEL;
    const __bf16* arow1 = arow0 + (size_t)16 * DMODEL;
    const __bf16* wbase = W + (size_t)(h * DK + ln) * DMODEL + 16 * hf;

    f32x8 acc[2][4] = {};
    KSlice s0, s1;
    load_slice(s0, arow0, arow1, wbase, 0, hf);

    // steady state: k = 32, 96, ..., 672 (11 double-iterations)
    for (int k = 32; k < DMODEL - 32; k += 64) {
        load_slice(s1, arow0, arow1, wbase, k, hf);        // produce s1
        compute_slice(acc, s0);                            // consume s0 (k-32)
        load_slice(s0, arow0, arow1, wbase, k + 32, hf);   // produce s0
        compute_slice(acc, s1);                            // consume s1 (k)
    }
    // epilogue: s0 holds k=704; load and consume k=736
    load_slice(s1, arow0, arow1, wbase, DMODEL - 32, hf);
    compute_slice(acc, s0);
    compute_slice(acc, s1);

    // C layout: row = r + 8*half, col = lane&15 (half-wave writes 32B runs)
#pragma unroll
    for (int mg = 0; mg < 2; ++mg)
#pragma unroll
        for (int nt = 0; nt < 4; ++nt)
#pragma unroll
            for (int r = 0; r < 8; ++r) {
                int t = m0 + mg * 16 + r + 8 * hf;
                int bb = t >> 11, s = t & (SEQ - 1);
                Out[(size_t)((bb * NHEAD + h) * SEQ + s) * DK + nt * 16 + ln] =
                    (__bf16)acc[mg][nt][r];
            }
}

// ---------------------------------------------------------------------------
// 2) RoPE in place on Q and K (bf16 [B,H,S,64]); pairs adjacent in dk.
// ---------------------------------------------------------------------------
__global__ __launch_bounds__(256) void rope_inplace(__bf16* __restrict__ Q,
                                                    __bf16* __restrict__ K) {
    int idx = blockIdx.x * 256 + threadIdx.x;
    const int total = BATCH * NHEAD * SEQ * (DK / 2);
    if (idx >= total) return;
    __bf16* buf = (blockIdx.y == 0) ? Q : K;
    int p = idx & 31;                 // pair index within head dim
    int s = (idx >> 5) & (SEQ - 1);   // sequence position
    float inv = __expf(-(float)p * (1.0f / 32.0f) * 9.210340371976184f); // theta^-2p/dk
    float ang = (float)s * inv;
    float sn, cs;
    __sincosf(ang, &sn, &cs);
    __bf16* e = buf + (size_t)idx * 2;
    float xe = (float)e[0], xo = (float)e[1];
    e[0] = (__bf16)(xe * cs - xo * sn);
    e[1] = (__bf16)(xe * sn + xo * cs);
}

// ---------------------------------------------------------------------------
// 3) V transpose [B,H,S,64] -> [B,H,64,S] via LDS so PV B-frags are contiguous
// ---------------------------------------------------------------------------
__global__ __launch_bounds__(256) void transpose_v(const __bf16* __restrict__ V,
                                                   __bf16* __restrict__ Vt) {
    __shared__ __bf16 tile[64][72];
    int s0 = blockIdx.x * 64, bh = blockIdx.y;
    const __bf16* src = V + (size_t)(bh * SEQ + s0) * DK;
#pragma unroll
    for (int it = 0; it < 16; ++it) {
        int lin = it * 256 + threadIdx.x;
        tile[lin >> 6][lin & 63] = src[(size_t)(lin >> 6) * DK + (lin & 63)];
    }
    __syncthreads();
    __bf16* dst = Vt + (size_t)bh * DK * SEQ + s0;
#pragma unroll
    for (int it = 0; it < 16; ++it) {
        int lin = it * 256 + threadIdx.x;
        int dl = lin >> 6, sl = lin & 63;
        dst[(size_t)dl * SEQ + sl] = tile[sl][dl];
    }
}

// ---------------------------------------------------------------------------
// 4) Flash attention: 4 waves x 16 query rows, key blocks of 64, causal skip.
//    Pipelined K-frag loads across nt steps; V-frag loads issued before the
//    softmax VALU chain so P*V WMMAs find data resident; next-block prefetch.
// ---------------------------------------------------------------------------
__global__ __launch_bounds__(128) void flash_attn(const __bf16* __restrict__ Qb,
                                                  const __bf16* __restrict__ Kb,
                                                  const __bf16* __restrict__ Vt,
                                                  __bf16* __restrict__ Ob) {
    constexpr float SCALE = 0.125f; // 1/sqrt(64)
    int qtile = blockIdx.x, h = blockIdx.y, bb = blockIdx.z;
    int wave = threadIdx.x >> 5, lane = threadIdx.x & 31;
    int ln = lane & 15, hf = lane >> 4;

    const __bf16* Qh = Qb + (size_t)((bb * NHEAD + h) * SEQ) * DK;
    const __bf16* Kh = Kb + (size_t)((bb * NHEAD + h) * SEQ) * DK;
    const __bf16* Vh = Vt + (size_t)((bb * NHEAD + h) * DK) * SEQ;
    int q0 = qtile * 64 + wave * 16;

    const __bf16* qrow = Qh + (size_t)(q0 + ln) * DK;
    bf16x16 qf0 = load_a_frag(qrow, hf);
    bf16x16 qf1 = load_a_frag(qrow + 32, hf);

    f32x8 o[4] = {};
    float m[8], l[8];
#pragma unroll
    for (int r = 0; r < 8; ++r) { m[r] = -3.0e38f; l[r] = 0.0f; }

    __shared__ __align__(16) __bf16 Plds[4][16][72]; // per-wave P tile, padded rows

    for (int kb = 0; kb <= qtile; ++kb) {
        int kbase = kb * 64;

        // L2 prefetch of next key/value block (global_prefetch_b8)
        if (kb < qtile) {
            __builtin_prefetch(Kh + (size_t)(kbase + 64 + lane) * DK, 0, 1);
            __builtin_prefetch(Kh + (size_t)(kbase + 96 + lane) * DK, 0, 1);
            __builtin_prefetch(Vh + (size_t)lane * SEQ + kbase + 64, 0, 1);
            __builtin_prefetch(Vh + (size_t)(lane + 32) * SEQ + kbase + 64, 0, 1);
        }

        // ---- scores: pipeline K-frag loads one nt ahead of the WMMAs ----
        f32x8 sc[4] = {};
        const __bf16* kp0 = Kh + (size_t)(kbase + ln) * DK + 16 * hf;
        bf16x16 k0 = load_b_frag(kp0);
        bf16x16 k1 = load_b_frag(kp0 + 32);
#pragma unroll
        for (int nt = 0; nt < 4; ++nt) {
            bf16x16 nk0 = k0, nk1 = k1;
            if (nt < 3) {
                const __bf16* kp = kp0 + (size_t)((nt + 1) * 16) * DK;
                nk0 = load_b_frag(kp);
                nk1 = load_b_frag(kp + 32);
            }
            sc[nt] = wmma_bf16(qf0, k0, sc[nt]);
            sc[nt] = wmma_bf16(qf1, k1, sc[nt]);
            k0 = nk0; k1 = nk1;
        }

        // ---- issue ALL V-fragment loads now; softmax hides their latency ----
        bf16x16 vf[4][2];
#pragma unroll
        for (int nt = 0; nt < 4; ++nt) {
            const __bf16* vp = Vh + (size_t)(nt * 16 + ln) * SEQ + kbase + 16 * hf;
            vf[nt][0] = load_b_frag(vp);
            vf[nt][1] = load_b_frag(vp + 32);
        }

        // ---- online softmax, per query row (rows split across lane halves) ----
#pragma unroll
        for (int r = 0; r < 8; ++r) {
            int qi = q0 + r + 8 * hf;
            float rmax = -3.0e38f;
#pragma unroll
            for (int nt = 0; nt < 4; ++nt) {
                int kc = kbase + nt * 16 + ln;
                float v = sc[nt][r] * SCALE;
                v = (kc <= qi) ? v : -3.0e38f;   // causal mask (finite -> no inf-inf NaN)
                sc[nt][r] = v;
                rmax = fmaxf(rmax, v);
            }
#pragma unroll
            for (int off = 1; off < 16; off <<= 1)
                rmax = fmaxf(rmax, __shfl_xor(rmax, off, 32));
            float mn = fmaxf(m[r], rmax);
            float corr = __expf(m[r] - mn);
            l[r] *= corr;
#pragma unroll
            for (int nt = 0; nt < 4; ++nt) o[nt][r] *= corr;
            float rsum = 0.0f;
#pragma unroll
            for (int nt = 0; nt < 4; ++nt) {
                float p = __expf(sc[nt][r] - mn);
                sc[nt][r] = p;
                rsum += p;
            }
#pragma unroll
            for (int off = 1; off < 16; off <<= 1)
                rsum += __shfl_xor(rsum, off, 32);
            l[r] += rsum;
            m[r] = mn;
        }

        // ---- restage P (C layout) through LDS into A-fragment layout ----
        __syncthreads();
#pragma unroll
        for (int nt = 0; nt < 4; ++nt)
#pragma unroll
            for (int r = 0; r < 8; ++r)
                Plds[wave][r + 8 * hf][nt * 16 + ln] = (__bf16)sc[nt][r];
        __syncthreads();
        const __bf16* prow = &Plds[wave][ln][0];
        bf16x16 pa0 = load_a_frag(prow, hf);
        bf16x16 pa1 = load_a_frag(prow + 32, hf);

        // ---- O += P * V, V fragments already resident ----
#pragma unroll
        for (int nt = 0; nt < 4; ++nt) {
            o[nt] = wmma_bf16(pa0, vf[nt][0], o[nt]);
            o[nt] = wmma_bf16(pa1, vf[nt][1], o[nt]);
        }
    }

    // normalize and write concat-head layout [B,S,D]
#pragma unroll
    for (int nt = 0; nt < 4; ++nt)
#pragma unroll
        for (int r = 0; r < 8; ++r) {
            int s = q0 + r + 8 * hf;
            float v = o[nt][r] / l[r];
            Ob[(size_t)(bb * SEQ + s) * DMODEL + h * DK + nt * 16 + ln] = (__bf16)v;
        }
}

// ---------------------------------------------------------------------------
// 5) Output projection: d_out[t, o] = A[t,:] . W_O[o,:], fp32 out.
//    Same ping-pong pipelined 32x64-per-wave NT GEMM.
// ---------------------------------------------------------------------------
__global__ __launch_bounds__(256) void gemm_out_proj(const __bf16* __restrict__ A,
                                                     const __bf16* __restrict__ W,
                                                     float* __restrict__ Out) {
    int wave = threadIdx.x >> 5, lane = threadIdx.x & 31;
    int ln = lane & 15, hf = lane >> 4;
    int m0 = blockIdx.x * 256 + wave * 32;
    int n0 = blockIdx.y * 64;

    const __bf16* arow0 = A + (size_t)(m0 + ln) * DMODEL;
    const __bf16* arow1 = arow0 + (size_t)16 * DMODEL;
    const __bf16* wbase = W + (size_t)(n0 + ln) * DMODEL + 16 * hf;

    f32x8 acc[2][4] = {};
    KSlice s0, s1;
    load_slice(s0, arow0, arow1, wbase, 0, hf);

    for (int k = 32; k < DMODEL - 32; k += 64) {
        load_slice(s1, arow0, arow1, wbase, k, hf);
        compute_slice(acc, s0);
        load_slice(s0, arow0, arow1, wbase, k + 32, hf);
        compute_slice(acc, s1);
    }
    load_slice(s1, arow0, arow1, wbase, DMODEL - 32, hf);
    compute_slice(acc, s0);
    compute_slice(acc, s1);

#pragma unroll
    for (int mg = 0; mg < 2; ++mg)
#pragma unroll
        for (int nt = 0; nt < 4; ++nt)
#pragma unroll
            for (int r = 0; r < 8; ++r) {
                int t = m0 + mg * 16 + r + 8 * hf;
                Out[(size_t)t * DMODEL + n0 + nt * 16 + ln] = acc[mg][nt][r];
            }
}

// ---------------------------------------------------------------------------
extern "C" void kernel_launch(void* const* d_in, const int* in_sizes, int n_in,
                              void* d_out, int out_size, void* d_ws, size_t ws_size,
                              hipStream_t stream) {
    (void)in_sizes; (void)n_in; (void)out_size; (void)ws_size;
    const float* X  = (const float*)d_in[0];
    const float* Wq = (const float*)d_in[1];
    const float* Wk = (const float*)d_in[2];
    const float* Wv = (const float*)d_in[3];
    const float* Wo = (const float*)d_in[4];

    char* ws = (char*)d_ws;
    size_t off = 0;
    auto carve = [&](size_t elems) {
        __bf16* p = (__bf16*)(ws + off);
        off += (elems * sizeof(__bf16) + 255) & ~(size_t)255;
        return p;
    };
    const size_t nX = (size_t)MTOT * DMODEL;     // 3,145,728
    const size_t nW = (size_t)DMODEL * DMODEL;   //   589,824
    __bf16* Xb  = carve(nX);
    __bf16* Wqb = carve(nW);
    __bf16* Wkb = carve(nW);
    __bf16* Wvb = carve(nW);
    __bf16* Wob = carve(nW);
    __bf16* Qb  = carve(nX);   // [B,H,S,64]
    __bf16* Kb  = carve(nX);
    __bf16* Vb  = carve(nX);
    __bf16* Vtb = carve(nX);   // [B,H,64,S]
    __bf16* Ab  = carve(nX);   // attention output [B,S,D]
    // total workspace ~40.5 MB

    cvt_f32_to_bf16<<<(int)(nX / 4 / 256), 256, 0, stream>>>(X,  Xb,  (int)nX);
    cvt_f32_to_bf16<<<(int)(nW / 4 / 256), 256, 0, stream>>>(Wq, Wqb, (int)nW);
    cvt_f32_to_bf16<<<(int)(nW / 4 / 256), 256, 0, stream>>>(Wk, Wkb, (int)nW);
    cvt_f32_to_bf16<<<(int)(nW / 4 / 256), 256, 0, stream>>>(Wv, Wvb, (int)nW);
    cvt_f32_to_bf16<<<(int)(nW / 4 / 256), 256, 0, stream>>>(Wo, Wob, (int)nW);

    dim3 gGemm(MTOT / 256, NHEAD);  // (16, 12)
    gemm_qkv_heads<<<gGemm, 256, 0, stream>>>(Xb, Wqb, Qb);
    gemm_qkv_heads<<<gGemm, 256, 0, stream>>>(Xb, Wkb, Kb);
    gemm_qkv_heads<<<gGemm, 256, 0, stream>>>(Xb, Wvb, Vb);

    int nPairs = BATCH * NHEAD * SEQ * (DK / 2);
    rope_inplace<<<dim3((nPairs + 255) / 256, 2), 256, 0, stream>>>(Qb, Kb);

    transpose_v<<<dim3(SEQ / 64, BATCH * NHEAD), 256, 0, stream>>>(Vb, Vtb);

    flash_attn<<<dim3(SEQ / 64, NHEAD, BATCH), 128, 0, stream>>>(Qb, Kb, Vtb, Ab);

    gemm_out_proj<<<gGemm, 256, 0, stream>>>(Ab, Wob, (float*)d_out);
}
